// Attention_22978075034026
// MI455X (gfx1250) — matrix operand order repaired and checked by reading, not compiled, not run on hardware
//
#include <hip/hip_runtime.h>
#include <hip/hip_bf16.h>

// ---------------------------------------------------------------------------
// Types for CDNA5 WMMA (wave32): v_wmma_f32_16x16x32_bf16
// ---------------------------------------------------------------------------
typedef __bf16 bf16_t;
typedef __attribute__((ext_vector_type(16))) __bf16 v16bf;
typedef __attribute__((ext_vector_type(8)))  __bf16 v8bf;   // 16-byte chunk
typedef __attribute__((ext_vector_type(8)))  float  v8f;

#define S_LEN 2048
#define DIM   4096
#define HQ    32
#define HKV   8
#define HD    128

union FragU { v16bf v; v8bf h[2]; };

// A fragment 16x32 (MxK), bf16, source row-major with leading dim ldr:
// lanes 0-15: M=lane, VGPR0-3 = K 0..7, VGPR4-7 = K 16..23
// lanes 16-31: M=lane-16, VGPR0-3 = K 8..15, VGPR4-7 = K 24..31
__device__ __forceinline__ v16bf load_frag_a(const bf16_t* base, int ldr, int lane) {
    const bf16_t* p = base + (size_t)(lane & 15) * ldr + ((lane >> 4) << 3);
    FragU u;
    u.h[0] = *(const v8bf*)(p);        // 16B
    u.h[1] = *(const v8bf*)(p + 16);   // 16B
    return u.v;
}

// B fragment 32x16 (KxN), bf16. Source layout: row n holds the K-dim
// contiguously (leading dim ldr). lanes 0-15: N=lane, K 0..7 and K 16..23;
// lanes 16-31: N=lane-16, K 8..15 and K 24..31.
__device__ __forceinline__ v16bf load_frag_b(const bf16_t* base, int ldr, int lane) {
    const bf16_t* p = base + (size_t)(lane & 15) * ldr + ((lane >> 4) << 3);
    FragU u;
    u.h[0] = *(const v8bf*)(p);        // K +0..7
    u.h[1] = *(const v8bf*)(p + 16);   // K +16..23
    return u.v;
}

__device__ __forceinline__ v8f wmma_bf16(v16bf a, v16bf b, v8f c) {
    return __builtin_amdgcn_wmma_f32_16x16x32_bf16(
        /*neg_a=*/false, a, /*neg_b=*/false, b,
        /*c_mod=*/(short)0, c, /*reuse_a=*/false, /*reuse_b=*/false);
}

// ---------------------------------------------------------------------------
// fp32 -> bf16 conversion
// ---------------------------------------------------------------------------
__global__ __launch_bounds__(256)
void cvt_f32_bf16(const float* __restrict__ src, bf16_t* __restrict__ dst, long n) {
    long i = ((long)blockIdx.x * blockDim.x + threadIdx.x) * 4;
    if (i + 3 < n) {
        float4 f = *(const float4*)(src + i);
        dst[i + 0] = (bf16_t)f.x;
        dst[i + 1] = (bf16_t)f.y;
        dst[i + 2] = (bf16_t)f.z;
        dst[i + 3] = (bf16_t)f.w;
    } else {
        for (; i < n; ++i) dst[i] = (bf16_t)src[i];
    }
}

// ---------------------------------------------------------------------------
// C[M,N] (f32) = A[M,K] (bf16, row-major) x B[N,K]^T (bf16, row-major)
// Block: 256 threads = 8 waves stacked in M; block tile 256(M) x 64(N).
// Each wave: 32(M) x 64(N) = 8 accumulators. Two-stage ping-pong software
// pipeline over K (step 64 = 2x32): each stage has its own named register
// set, loaded in one position and consumed in the other, so the compiler
// rotates registers instead of emitting v_mov copies, and one stage's
// 12 b128 loads stay in flight across the other stage's 8 WMMAs.
// ---------------------------------------------------------------------------
__global__ __launch_bounds__(256)
void gemm_bf16_nt(const bf16_t* __restrict__ A, const bf16_t* __restrict__ B,
                  float* __restrict__ C, int M, int N, int K) {
    const int lane = threadIdx.x & 31;
    const int wave = threadIdx.x >> 5;
    const int m0 = blockIdx.y * 256 + wave * 32;
    const int n0 = blockIdx.x * 64;

    const bf16_t* A0 = A + (size_t)m0 * K;
    const bf16_t* A1 = A0 + (size_t)16 * K;
    const bf16_t* Bb = B + (size_t)n0 * K;

    v8f acc[2][4] = {};

    v16bf a0X, a1X, bX[4];   // stage X
    v16bf a0Y, a1Y, bY[4];   // stage Y

    // prologue: stage X <- k = 0
    a0X = load_frag_a(A0, K, lane);
    a1X = load_frag_a(A1, K, lane);
#pragma unroll
    for (int t = 0; t < 4; ++t)
        bX[t] = load_frag_b(Bb + (size_t)t * 16 * K, K, lane);

    int k0 = 0;
    for (; k0 + 64 < K; k0 += 64) {
        // stage Y <- k0+32 (loads overlap stage X compute)
        a0Y = load_frag_a(A0 + k0 + 32, K, lane);
        a1Y = load_frag_a(A1 + k0 + 32, K, lane);
#pragma unroll
        for (int t = 0; t < 4; ++t)
            bY[t] = load_frag_b(Bb + (size_t)t * 16 * K + k0 + 32, K, lane);
        __builtin_prefetch(A0 + k0 + 1024, 0, 1);   // global_prefetch_b8
        // compute stage X (k0)
#pragma unroll
        for (int t = 0; t < 4; ++t) {
            acc[0][t] = wmma_bf16(a0X, bX[t], acc[0][t]);
            acc[1][t] = wmma_bf16(a1X, bX[t], acc[1][t]);
        }
        // stage X <- k0+64 (loads overlap stage Y compute)
        a0X = load_frag_a(A0 + k0 + 64, K, lane);
        a1X = load_frag_a(A1 + k0 + 64, K, lane);
#pragma unroll
        for (int t = 0; t < 4; ++t)
            bX[t] = load_frag_b(Bb + (size_t)t * 16 * K + k0 + 64, K, lane);
        __builtin_prefetch(Bb + (size_t)16 * K + k0 + 1024, 0, 1);
        // compute stage Y (k0+32)
#pragma unroll
        for (int t = 0; t < 4; ++t) {
            acc[0][t] = wmma_bf16(a0Y, bY[t], acc[0][t]);
            acc[1][t] = wmma_bf16(a1Y, bY[t], acc[1][t]);
        }
    }
    // epilogue: k0 == K-64; stage X holds k0. Load stage Y <- k0+32, finish.
    a0Y = load_frag_a(A0 + k0 + 32, K, lane);
    a1Y = load_frag_a(A1 + k0 + 32, K, lane);
#pragma unroll
    for (int t = 0; t < 4; ++t)
        bY[t] = load_frag_b(Bb + (size_t)t * 16 * K + k0 + 32, K, lane);
#pragma unroll
    for (int t = 0; t < 4; ++t) {
        acc[0][t] = wmma_bf16(a0X, bX[t], acc[0][t]);
        acc[1][t] = wmma_bf16(a1X, bX[t], acc[1][t]);
    }
#pragma unroll
    for (int t = 0; t < 4; ++t) {
        acc[0][t] = wmma_bf16(a0Y, bY[t], acc[0][t]);
        acc[1][t] = wmma_bf16(a1Y, bY[t], acc[1][t]);
    }

    // C layout: lane&15 = column within tile, VGPR r = row r + 8*(lane>>4)
    const int nc = n0 + (lane & 15);
#pragma unroll
    for (int s = 0; s < 2; ++s) {
        const int mr = m0 + s * 16 + ((lane >> 4) << 3);
#pragma unroll
        for (int t = 0; t < 4; ++t)
#pragma unroll
            for (int r = 0; r < 8; ++r)
                C[(size_t)(mr + r) * N + nc + t * 16] = acc[s][t][r];
    }
}

// ---------------------------------------------------------------------------
// RoPE + pack fp32 [S][nheads*HD] -> bf16 [nheads][S][HD]
// ---------------------------------------------------------------------------
__global__ __launch_bounds__(64)
void rope_pack(const float* __restrict__ src, const float* __restrict__ cosb,
               const float* __restrict__ sinb, bf16_t* __restrict__ dst, int nheads) {
    const int s = blockIdx.x, h = blockIdx.y, i = threadIdx.x;  // i in [0,64)
    const float* p = src + ((size_t)s * nheads + h) * HD;
    float x1 = p[2 * i], x2 = p[2 * i + 1];
    float c = cosb[s * (HD / 2) + i], sn = sinb[s * (HD / 2) + i];
    bf16_t* q = dst + ((size_t)h * S_LEN + s) * HD;
    q[2 * i]     = (bf16_t)(x1 * c - x2 * sn);
    q[2 * i + 1] = (bf16_t)(x1 * sn + x2 * c);
}

// V: fp32 [S][HKV*HD] -> bf16 [HKV][HD][S]   (transposed for P@V B-fragments)
__global__ __launch_bounds__(128)
void v_pack_t(const float* __restrict__ src, bf16_t* __restrict__ dst) {
    const int s = blockIdx.x, h = blockIdx.y, d = threadIdx.x;  // d in [0,128)
    dst[((size_t)h * HD + d) * S_LEN + s] =
        (bf16_t)src[((size_t)s * HKV + h) * HD + d];
}

// ---------------------------------------------------------------------------
// Flash attention, causal + window. One block = 1 Q head x 128 query rows.
// 8 waves, each wave owns 16 query rows. bf16 WMMA for QK^T and P@V,
// fp32 online softmax held in the WMMA C layout. K and V fragments for the
// whole 32-key step are issued up front so V loads overlap the softmax VALU
// section and the LDS round-trip of P.
// ---------------------------------------------------------------------------
__global__ __launch_bounds__(256)
void flash_attn(const bf16_t* __restrict__ qb,   // [HQ][S][HD]
                const bf16_t* __restrict__ kb,   // [HKV][S][HD]
                const bf16_t* __restrict__ vt,   // [HKV][HD][S]
                bf16_t* __restrict__ ob,         // [S][HQ*HD]
                const int* __restrict__ winp) {
    __shared__ bf16_t lds_p[8 * 16 * 32];        // per-wave 16x32 P tile

    const int lane = threadIdx.x & 31;
    const int wave = threadIdx.x >> 5;
    const int half = lane >> 4;
    const int ln   = lane & 15;
    const int h    = blockIdx.y;
    const int hk   = h >> 2;                     // HQ/HKV = 4
    const int s0   = blockIdx.x * 128 + wave * 16;
    const int window = *winp;
    const float scale = 0.08838834764831845f;    // 1/sqrt(128)

    // Q fragments for this wave's 16 rows: 4 K-chunks of 32 over HD=128
    v16bf qf[4];
#pragma unroll
    for (int c = 0; c < 4; ++c)
        qf[c] = load_frag_a(qb + ((size_t)h * S_LEN + s0) * HD + c * 32, HD, lane);

    v8f o[8] = {};                               // 16 rows x 128 cols (8 d-tiles)
    float mrow[8], lrow[8];
#pragma unroll
    for (int r = 0; r < 8; ++r) { mrow[r] = -3.0e38f; lrow[r] = 0.0f; }

    bf16_t* pbuf = lds_p + wave * (16 * 32);
    const int kmax = blockIdx.x * 128 + 127;     // block-uniform causal bound
    const bf16_t* krow = kb + (size_t)hk * S_LEN * HD;
    const bf16_t* vrow = vt + (size_t)hk * HD * S_LEN;

    for (int j0 = 0; j0 <= kmax; j0 += 32) {
        // ---- issue ALL loads for this key block up front --------------
        __builtin_prefetch(krow + (size_t)(j0 + 32) * HD, 0, 1);
        v16bf kfr[2][4];
#pragma unroll
        for (int t = 0; t < 2; ++t)
#pragma unroll
            for (int c = 0; c < 4; ++c)
                kfr[t][c] = load_frag_b(krow + (size_t)(j0 + t * 16) * HD + c * 32,
                                        HD, lane);
        v16bf vfr[8];
#pragma unroll
        for (int t = 0; t < 8; ++t)
            vfr[t] = load_frag_b(vrow + (size_t)(t * 16) * S_LEN + j0, S_LEN, lane);

        // ---- scores for 32 keys: two 16x16 WMMA tiles -----------------
        float pr[2][8];
        float tmax[8];
#pragma unroll
        for (int r = 0; r < 8; ++r) tmax[r] = -3.0e38f;
#pragma unroll
        for (int t = 0; t < 2; ++t) {
            v8f sacc = {};
#pragma unroll
            for (int c = 0; c < 4; ++c)
                sacc = wmma_bf16(qf[c], kfr[t][c], sacc);
            const int j = j0 + t * 16 + ln;
#pragma unroll
            for (int r = 0; r < 8; ++r) {
                const int i = s0 + r + (half << 3);
                float sv = sacc[r] * scale;
                bool ok = (j <= i) && ((i - j) < window);
                sv = ok ? sv : -1.0e30f;
                pr[t][r] = sv;
                tmax[r] = fmaxf(tmax[r], sv);
            }
        }
        // ---- row max across the 16 lanes holding each row -------------
#pragma unroll
        for (int r = 0; r < 8; ++r) {
            float v = tmax[r];
            v = fmaxf(v, __shfl_xor(v, 1));
            v = fmaxf(v, __shfl_xor(v, 2));
            v = fmaxf(v, __shfl_xor(v, 4));
            v = fmaxf(v, __shfl_xor(v, 8));
            tmax[r] = v;
        }
        // ---- online softmax update; stage P (bf16) into LDS -----------
        asm volatile("" ::: "memory");
#pragma unroll
        for (int r = 0; r < 8; ++r) {
            float mnew  = fmaxf(mrow[r], tmax[r]);
            float alpha = __expf(mrow[r] - mnew);
            float p0 = __expf(pr[0][r] - mnew);
            float p1 = __expf(pr[1][r] - mnew);
            pbuf[(r + (half << 3)) * 32 + ln]      = (bf16_t)p0;
            pbuf[(r + (half << 3)) * 32 + 16 + ln] = (bf16_t)p1;
            float ps = p0 + p1;
            ps += __shfl_xor(ps, 1);
            ps += __shfl_xor(ps, 2);
            ps += __shfl_xor(ps, 4);
            ps += __shfl_xor(ps, 8);
            lrow[r] = lrow[r] * alpha + ps;
            mrow[r] = mnew;
#pragma unroll
            for (int t = 0; t < 8; ++t) o[t][r] *= alpha;
        }
        // Wave-private LDS tile; DS ops are in-order per wave. Fence the
        // compiler + hardware before re-reading as an A-fragment.
        asm volatile("s_wait_dscnt 0" ::: "memory");
        // ---- O += P(16x32) @ V(32x128) --------------------------------
        v16bf pa = load_frag_a(pbuf, 32, lane);
#pragma unroll
        for (int t = 0; t < 8; ++t)
            o[t] = wmma_bf16(pa, vfr[t], o[t]);
    }
    // ---- normalize, store bf16 [S][HQ*HD] for the Wo GEMM -------------
#pragma unroll
    for (int r = 0; r < 8; ++r) {
        float inv = 1.0f / lrow[r];
        const size_t srow =
            (size_t)(s0 + r + (half << 3)) * (HQ * HD) + (size_t)h * HD + ln;
#pragma unroll
        for (int t = 0; t < 8; ++t)
            ob[srow + t * 16] = (bf16_t)(o[t][r] * inv);
    }
}

// ---------------------------------------------------------------------------
// Host side
// ---------------------------------------------------------------------------
extern "C" void kernel_launch(void* const* d_in, const int* in_sizes, int n_in,
                              void* d_out, int out_size, void* d_ws, size_t ws_size,
                              hipStream_t stream) {
    const float* x    = (const float*)d_in[0];
    const float* wq   = (const float*)d_in[1];
    const float* wk   = (const float*)d_in[2];
    const float* wv   = (const float*)d_in[3];
    const float* wo   = (const float*)d_in[4];
    const float* cosb = (const float*)d_in[5];
    const float* sinb = (const float*)d_in[6];
    const int*   win  = (const int*)d_in[7];
    float* out = (float*)d_out;

    char* ws = (char*)d_ws;
    auto alloc = [&](size_t bytes) -> char* {
        char* p = ws;
        ws += (bytes + 255) & ~(size_t)255;
        return p;
    };
    bf16_t* xb  = (bf16_t*)alloc((size_t)S_LEN * DIM * 2);
    bf16_t* wqb = (bf16_t*)alloc((size_t)DIM * DIM * 2);
    bf16_t* wkb = (bf16_t*)alloc((size_t)(HKV * HD) * DIM * 2);
    bf16_t* wvb = (bf16_t*)alloc((size_t)(HKV * HD) * DIM * 2);
    bf16_t* wob = (bf16_t*)alloc((size_t)DIM * DIM * 2);
    float*  qf  = (float*)alloc((size_t)S_LEN * DIM * 4);
    float*  kf  = (float*)alloc((size_t)S_LEN * (HKV * HD) * 4);
    float*  vf  = (float*)alloc((size_t)S_LEN * (HKV * HD) * 4);
    bf16_t* qpk = (bf16_t*)alloc((size_t)HQ * S_LEN * HD * 2);
    bf16_t* kpk = (bf16_t*)alloc((size_t)HKV * S_LEN * HD * 2);
    bf16_t* vpk = (bf16_t*)alloc((size_t)HKV * HD * S_LEN * 2);
    // attention output (bf16) aliases the qf scratch (qf is dead by then)
    bf16_t* apk = (bf16_t*)qf;

    auto cvt = [&](const float* s, bf16_t* d, long n) {
        long quads = n / 4;
        int thr = 256;
        cvt_f32_bf16<<<(unsigned)((quads + thr - 1) / thr), thr, 0, stream>>>(s, d, n);
    };
    cvt(x,  xb,  (long)S_LEN * DIM);
    cvt(wq, wqb, (long)DIM * DIM);
    cvt(wk, wkb, (long)(HKV * HD) * DIM);
    cvt(wv, wvb, (long)(HKV * HD) * DIM);
    cvt(wo, wob, (long)DIM * DIM);

    dim3 blk(256);
    // Q/K/V projections (fp32 out, RoPE still to apply)
    gemm_bf16_nt<<<dim3(DIM / 64, S_LEN / 256), blk, 0, stream>>>(
        xb, wqb, qf, S_LEN, DIM, DIM);
    gemm_bf16_nt<<<dim3((HKV * HD) / 64, S_LEN / 256), blk, 0, stream>>>(
        xb, wkb, kf, S_LEN, HKV * HD, DIM);
    gemm_bf16_nt<<<dim3((HKV * HD) / 64, S_LEN / 256), blk, 0, stream>>>(
        xb, wvb, vf, S_LEN, HKV * HD, DIM);

    // RoPE + layout packing
    rope_pack<<<dim3(S_LEN, HQ),  64, 0, stream>>>(qf, cosb, sinb, qpk, HQ);
    rope_pack<<<dim3(S_LEN, HKV), 64, 0, stream>>>(kf, cosb, sinb, kpk, HKV);
    v_pack_t<<<dim3(S_LEN, HKV), 128, 0, stream>>>(vf, vpk);

    // Attention
    flash_attn<<<dim3(S_LEN / 128, HQ), 256, 0, stream>>>(qpk, kpk, vpk, apk, win);

    // Output projection -> fp32 d_out
    gemm_bf16_nt<<<dim3(DIM / 64, S_LEN / 256), blk, 0, stream>>>(
        apk, wob, out, S_LEN, DIM, DIM);
}
